// SLC2_65377992179757
// MI455X (gfx1250) — matrix operand order repaired and checked by reading, not compiled, not run on hardware
//
#include <hip/hip_runtime.h>
#include <stdint.h>

// ---------- types ----------
typedef __bf16 bf16;
typedef __attribute__((ext_vector_type(16))) __bf16 bf16x16;
typedef __attribute__((ext_vector_type(4)))  __bf16 bf16x4;
typedef __attribute__((ext_vector_type(8)))  float  f32x8;
typedef __attribute__((ext_vector_type(4)))  unsigned int u32x4;

union FragU { bf16x16 v; u32x4 q[2]; };

// ---------- problem constants ----------
#define GM   16384L   // B*S
#define GE   2048     // E (also K of every GEMM)
#define GH   16
#define GDH  128
#define GS   4096     // S (power of two)
#define GKC  4        // conv taps

// ---------- GEMM tiling ----------
#define TM   128
#define TN   128
#define TK   32
#define LDSS 40       // LDS row stride in bf16 elems (80B: 16B aligned, conflict-light)

// ---------- CDNA5 async copy to LDS (ASYNCcnt path) ----------
__device__ __forceinline__ void slc2_async_b128(unsigned ldsOff, const void* gaddr) {
    // GLOBAL_LOAD_ASYNC_TO_LDS_B128, GV mode (saddr = off), per-lane 64-bit address.
    asm volatile("global_load_async_to_lds_b128 %0, %1, off"
                 :
                 : "v"(ldsOff), "v"(gaddr)
                 : "memory");
}
__device__ __forceinline__ void slc2_wait_async0() {
    asm volatile("s_wait_asynccnt 0x0" ::: "memory");
}
// flat address of a __shared__ object: low 32 bits == LDS byte offset (ISA §10.2)
__device__ __forceinline__ unsigned slc2_lds_off(const void* p) {
    return (unsigned)(size_t)p;
}

// =====================================================================
// fp32 -> bf16 convert (n must be a multiple of 4; all our sizes are)
// =====================================================================
__global__ __launch_bounds__(256) void slc2_cvt_bf16(const float* __restrict__ in,
                                                     bf16* __restrict__ out, long n4) {
    long i = (long)blockIdx.x * 256 + threadIdx.x;
    if (i < n4) {
        float4 v = ((const float4*)in)[i];
        bf16x4 o;
        o.x = (bf16)v.x; o.y = (bf16)v.y; o.z = (bf16)v.z; o.w = (bf16)v.w;
        ((bf16x4*)out)[i] = o;
    }
}

// =====================================================================
// C[M,N] (fp32) = A[M,2048] (bf16, row-major) * W[N,2048]^T (bf16, row-major)
// 128x128 block tile, 8 wave32 waves, each wave = 32x64 (2x4 wmma tiles).
// Double-buffered LDS filled by global_load_async_to_lds_b128; one barrier
// per K-step (async issue for tile k+1 happens after the barrier that proves
// every wave finished reading that buffer in iteration k-1).
// =====================================================================
__global__ __launch_bounds__(256) void slc2_gemm_bf16_nt(const bf16* __restrict__ A,
                                                         const bf16* __restrict__ W,
                                                         float* __restrict__ C,
                                                         int N) {
    __shared__ bf16 sA[2][TM * LDSS];
    __shared__ bf16 sB[2][TM * LDSS];

    const int tid  = threadIdx.x;
    const int lane = tid & 31;
    const int wid  = tid >> 5;     // 0..7
    const int wm   = wid >> 1;     // 0..3 -> M offset wm*32
    const int wn   = wid & 1;      // 0..1 -> N offset wn*64
    const long bm  = (long)blockIdx.y * TM;
    const long bn  = (long)blockIdx.x * TN;

    f32x8 acc[2][4];
#pragma unroll
    for (int i = 0; i < 2; ++i)
#pragma unroll
        for (int j = 0; j < 4; ++j)
            acc[i][j] = (f32x8){0.f, 0.f, 0.f, 0.f, 0.f, 0.f, 0.f, 0.f};

    // global->LDS: thread t owns 16 bf16 (32B = two b128 chunks) of one tile row
    const int  lr = tid >> 1;            // 0..127
    const int  lk = (tid & 1) << 4;      // 0 or 16
    const bf16* aPtr = A + (bm + lr) * (long)GE + lk;
    const bf16* wPtr = W + (bn + lr) * (long)GE + lk;
    const bool bValid = (bn + lr) < N;

    const unsigned ldsA0 = slc2_lds_off(&sA[0][lr * LDSS + lk]);
    const unsigned ldsA1 = slc2_lds_off(&sA[1][lr * LDSS + lk]);
    const unsigned ldsB0 = slc2_lds_off(&sB[0][lr * LDSS + lk]);
    const unsigned ldsB1 = slc2_lds_off(&sB[1][lr * LDSS + lk]);

    // prologue: tile 0 -> buffer 0
    slc2_async_b128(ldsA0, aPtr);
    slc2_async_b128(ldsA0 + 16, aPtr + 8);
    if (bValid) {
        slc2_async_b128(ldsB0, wPtr);
        slc2_async_b128(ldsB0 + 16, wPtr + 8);
    }
    aPtr += TK; wPtr += TK;

    const int fm  = lane & 15;            // fragment row/col within 16
    const int kbA = (lane >> 4) << 3;     // A-frag: K base 0/8 (ISA 16-bit A layout)
    const int kbB = (lane >> 4) << 3;     // B-frag: K base 0/8

    const int NK = GE / TK;               // 64
    for (int kt = 0; kt < NK; ++kt) {
        slc2_wait_async0();   // tile kt resident in LDS (this wave's portion)
        __syncthreads();      // ... and everyone else's; also releases buf (kt+1)&1

        const int cur = kt & 1;
        if (kt + 1 < NK) {
            const unsigned la = cur ? ldsA0 : ldsA1;
            const unsigned lb = cur ? ldsB0 : ldsB1;
            slc2_async_b128(la, aPtr);
            slc2_async_b128(la + 16, aPtr + 8);
            if (bValid) {
                slc2_async_b128(lb, wPtr);
                slc2_async_b128(lb + 16, wPtr + 8);
            }
            aPtr += TK; wPtr += TK;
        }

        const bf16* cA = sA[cur];
        const bf16* cB = sB[cur];

        bf16x16 af[2], bfg[4];
#pragma unroll
        for (int i = 0; i < 2; ++i) {
            const bf16* p = &cA[(wm * 32 + i * 16 + fm) * LDSS + kbA];
            FragU u;
            u.q[0] = *(const u32x4*)(p);        // elems 0..7  : K = kb+0..7
            u.q[1] = *(const u32x4*)(p + 16);   // elems 8..15 : K = kb+16..23
            af[i] = u.v;
        }
#pragma unroll
        for (int j = 0; j < 4; ++j) {
            const bf16* p = &cB[(wn * 64 + j * 16 + fm) * LDSS + kbB];
            FragU u;
            u.q[0] = *(const u32x4*)(p);        // K = kb+0..7
            u.q[1] = *(const u32x4*)(p + 16);   // K = kb+16..23
            bfg[j] = u.v;
        }

#pragma unroll
        for (int i = 0; i < 2; ++i)
#pragma unroll
            for (int j = 0; j < 4; ++j)
                acc[i][j] = __builtin_amdgcn_wmma_f32_16x16x32_bf16(
                    false, af[i], false, bfg[j], (short)0, acc[i][j], false, false);
    }

    // epilogue: lane l, vgpr r -> row = base + (l>>4)*8 + r, col = base + (l&15)
    const long rowBase = bm + wm * 32 + ((lane >> 4) << 3);
    const int  colBase = (int)bn + wn * 64 + (lane & 15);
#pragma unroll
    for (int i = 0; i < 2; ++i) {
#pragma unroll
        for (int j = 0; j < 4; ++j) {
            int col = colBase + j * 16;
            if (col < N) {
#pragma unroll
                for (int r = 0; r < 8; ++r) {
                    long row = rowBase + i * 16 + r;
                    C[row * (long)N + col] = acc[i][j][r];
                }
            }
        }
    }
}

// =====================================================================
// conv_in = silu(A * silu(alpha)) * x   (in-place over x)
// =====================================================================
__device__ __forceinline__ float slc2_silu(float v) {
    return v * (1.0f / (1.0f + __expf(-v)));
}

__global__ __launch_bounds__(256) void slc2_conv_in(float* __restrict__ x,
                                                    const float* __restrict__ Abuf,
                                                    const float* __restrict__ alpha) {
    long idx = (long)blockIdx.x * 256 + threadIdx.x;
    long row = idx >> 11;           // / 2048
    int  e   = (int)(idx & 2047);
    int  h   = e >> 7;              // / 128
    int  d   = e & 127;
    float sa  = slc2_silu(alpha[row * GH + h]);
    float af  = Abuf[row * GDH + d] * sa;
    x[idx] = slc2_silu(af) * x[idx];
}

// =====================================================================
// xAB = Bg * silu(depthwise_causal_conv(conv_in))   (in-place over Bg)
// =====================================================================
__global__ __launch_bounds__(256) void slc2_conv_gate(const float* __restrict__ ci,
                                                      const float* __restrict__ w,
                                                      float* __restrict__ bg) {
    long idx = (long)blockIdx.x * 256 + threadIdx.x;
    long row = idx >> 11;
    int  e   = (int)(idx & 2047);
    int  s   = (int)(row & (GS - 1));   // position within batch
    float acc = 0.f;
#pragma unroll
    for (int k = 0; k < GKC; ++k) {
        int sp = s + k - (GKC - 1);
        if (sp >= 0)
            acc += w[e * GKC + k] * ci[idx + ((long)(k - (GKC - 1)) << 11)];
    }
    bg[idx] = bg[idx] * slc2_silu(acc);
}

// =====================================================================
// per-row RMS norm, fp32 in -> bf16 out (one block per row of 2048)
// =====================================================================
__global__ __launch_bounds__(256) void slc2_rms_bf16(const float* __restrict__ xab,
                                                     bf16* __restrict__ out) {
    __shared__ float red[256];
    const long row = blockIdx.x;
    const float* p = xab + row * (long)GE;
    float s = 0.f;
#pragma unroll
    for (int i = 0; i < 8; ++i) {
        float v = p[threadIdx.x + i * 256];
        s += v * v;
    }
    red[threadIdx.x] = s;
    __syncthreads();
    for (int off = 128; off > 0; off >>= 1) {
        if (threadIdx.x < off) red[threadIdx.x] += red[threadIdx.x + off];
        __syncthreads();
    }
    const float scale = rsqrtf(red[0] * (1.0f / (float)GE) + 1.1920929e-07f);
    bf16* o = out + row * (long)GE;
#pragma unroll
    for (int i = 0; i < 8; ++i) {
        int e = threadIdx.x + i * 256;
        o[e] = (bf16)(p[e] * scale);
    }
}

// =====================================================================
// launch
// =====================================================================
extern "C" void kernel_launch(void* const* d_in, const int* in_sizes, int n_in,
                              void* d_out, int out_size, void* d_ws, size_t ws_size,
                              hipStream_t stream) {
    const float* hs     = (const float*)d_in[0];
    const float* Wx     = (const float*)d_in[1];
    const float* Walpha = (const float*)d_in[2];
    const float* WA     = (const float*)d_in[3];
    const float* WB     = (const float*)d_in[4];
    const float* convw  = (const float*)d_in[5];
    const float* Wc     = (const float*)d_in[6];
    float* out = (float*)d_out;

    const long M = GM, E = GE;

    // workspace carve-out (~372 MB)
    char* ws = (char*)d_ws;
    size_t o = 0;
    auto take = [&](size_t bytes) {
        size_t r = o;
        o = (o + bytes + 255) & ~(size_t)255;
        return r;
    };
    bf16* hs_b  = (bf16*)(ws + take(M * E * 2));        // later reused for rmsnorm output
    bf16* Wx_b  = (bf16*)(ws + take((size_t)E * E * 2));
    bf16* Wal_b = (bf16*)(ws + take((size_t)GH * E * 2));
    bf16* WA_b  = (bf16*)(ws + take((size_t)GDH * E * 2));
    bf16* WB_b  = (bf16*)(ws + take((size_t)E * E * 2));
    bf16* Wc_b  = (bf16*)(ws + take((size_t)E * E * 2));
    float* xbuf = (float*)(ws + take(M * E * 4));       // x -> conv_in (in-place)
    float* bgbuf = (float*)(ws + take(M * E * 4));      // Bg -> xAB (in-place)
    float* Abuf  = (float*)(ws + take(M * GDH * 4));
    float* albuf = (float*)(ws + take(M * GH * 4));
    (void)ws_size; (void)in_sizes; (void)n_in; (void)out_size;

    dim3 blk(256);

    // fp32 -> bf16 staging (vectorized, counts are multiples of 4)
    auto cvt = [&](const float* src, bf16* dst, long n) {
        long n4 = n >> 2;
        slc2_cvt_bf16<<<(unsigned)((n4 + 255) / 256), blk, 0, stream>>>(src, dst, n4);
    };
    cvt(hs, hs_b, M * E);
    cvt(Wx, Wx_b, (long)E * E);
    cvt(Walpha, Wal_b, (long)GH * E);
    cvt(WA, WA_b, (long)GDH * E);
    cvt(WB, WB_b, (long)E * E);
    cvt(Wc, Wc_b, (long)E * E);

    // four projections (WMMA bf16, f32 accumulate)
    dim3 gBig(E / TN, (unsigned)(M / TM));
    slc2_gemm_bf16_nt<<<gBig, blk, 0, stream>>>(hs_b, Wx_b, xbuf, (int)E);
    slc2_gemm_bf16_nt<<<gBig, blk, 0, stream>>>(hs_b, WB_b, bgbuf, (int)E);
    slc2_gemm_bf16_nt<<<dim3(1, (unsigned)(M / TM)), blk, 0, stream>>>(hs_b, WA_b, Abuf, GDH);
    slc2_gemm_bf16_nt<<<dim3(1, (unsigned)(M / TM)), blk, 0, stream>>>(hs_b, Wal_b, albuf, GH);

    // elementwise chain
    unsigned gElem = (unsigned)(M * E / 256);
    slc2_conv_in<<<gElem, blk, 0, stream>>>(xbuf, Abuf, albuf);
    slc2_conv_gate<<<gElem, blk, 0, stream>>>(xbuf, convw, bgbuf);
    slc2_rms_bf16<<<(unsigned)M, blk, 0, stream>>>(bgbuf, hs_b);

    // final projection -> d_out
    slc2_gemm_bf16_nt<<<gBig, blk, 0, stream>>>(hs_b, Wc_b, out, (int)E);
}
